// IntegralDiffAttention_75368086110258
// MI455X (gfx1250) — matrix operand order repaired and checked by reading, not compiled, not run on hardware
//
#include <hip/hip_runtime.h>

typedef __attribute__((ext_vector_type(16))) _Float16 v16h;
typedef __attribute__((ext_vector_type(8)))  _Float16 v8h;
typedef __attribute__((ext_vector_type(4)))  _Float16 v4h;
typedef __attribute__((ext_vector_type(2)))  _Float16 v2h;
typedef __attribute__((ext_vector_type(8)))  float    v8f;
typedef __attribute__((ext_vector_type(4)))  float    v4f;

// Problem constants (fixed by the reference)
// B=2, N=M=1024, C=1024, H=16, S=8, P=4, HD=64, SD=8
#define ROWS_TOT 2048     // B*N flattened rows for projections
#define CD       1024
#define NH       16
#define HDIM     64
#define COEF_BASE 0.08838834764831845f   // (1/sqrt(8)) / P

__device__ __forceinline__ v16h make_v16h(v8h a, v8h b) {
    return __builtin_shufflevector(a, b, 0,1,2,3,4,5,6,7,8,9,10,11,12,13,14,15);
}
__device__ __forceinline__ v8f wmma_f16(v16h a, v16h b, v8f c) {
    return __builtin_amdgcn_wmma_f32_16x16x32_f16(false, a, false, b, (short)0, c, false, false);
}

// LDS byte address (wave-relative) of a __shared__ object, for inline-asm VMEM->LDS ops
__device__ __forceinline__ unsigned lds_addr_of(const void* p) {
    return (unsigned)(size_t)(__attribute__((address_space(3))) const void*)p;
}
// CDNA5 async global->LDS copy of 16 bytes per lane (tracked by ASYNCcnt)
__device__ __forceinline__ void async_copy_b128(unsigned lds_byte_addr, const void* gaddr) {
    asm volatile("global_load_async_to_lds_b128 %0, %1, off"
                 :: "v"(lds_byte_addr), "v"(gaddr)
                 : "memory");
}
__device__ __forceinline__ void wait_asynccnt0() {
    asm volatile("s_wait_asynccnt 0x0" ::: "memory");
}

// ---------------------------------------------------------------------------
// Tiled GEMM:  D[row, col] = A[row, :] @ W[:, col] + bias[col]
// A: [2048 x 1024] (f32 input or f16 workspace), W: [1024 x 1024] f32 (in,out)
// Block tile 64 rows x 128 cols, 256 threads = 8 waves, each wave: 4 16x16 tiles.
// Double-buffered LDS, one barrier per K-iteration; f16 A-tiles staged with
// global_load_async_to_lds_b128.
// MODE 0: Q -> scale by lambda coefs, store f16 head-major [bh][n][d]
// MODE 1: K -> store f16 head-major [bh][m][d]
// MODE 2: V -> store f16 transposed  [bh][d][m]
// MODE 3: O -> store f32 row-major to d_out
// ---------------------------------------------------------------------------
template <int MODE, bool AISHALF>
__global__ __launch_bounds__(256)
void idattn_gemm_kernel(const void* __restrict__ Aptr,
                        const float* __restrict__ W,
                        const float* __restrict__ bias,
                        const float* __restrict__ lambda_log,
                        _Float16* __restrict__ dstH,
                        float* __restrict__ dstF)
{
    __shared__ alignas(32) _Float16 sA[2][64 * 48];    // 64 rows x 32 K (stride 48)
    __shared__ alignas(32) _Float16 sB[2][128 * 48];   // 128 cols x 32 K (transposed W)

    const int tid  = threadIdx.x;
    const int wave = tid >> 5;
    const int lane = tid & 31;
    const int lhi  = lane >> 4;    // 0/1 half-wave
    const int llo  = lane & 15;

    const int rowblk = blockIdx.y * 64;
    const int colblk = blockIdx.x * 128;

    const int rt    = wave >> 1;          // row tile 0..3
    const int cbase = (wave & 1) * 64;    // col sub-block

    const float*    Af = (const float*)Aptr;
    const _Float16* Ah = (const _Float16*)Aptr;

    // Per-thread staging coordinates
    const int ar  = tid >> 2;             // async A: row 0..63
    const int ac8 = (tid & 3) * 8;        // async A: 8-half chunk
    v4f  aRegF[2];                        // f32 A staging regs

    float bReg0[8], bReg1[8];             // W staging regs (k-pairs)

    v8f acc[4] = {};

    // ---- staging helpers (inlined manually via lambdas) ----
    auto loadA = [&](int k0) {
        if constexpr (AISHALF) {
            // async copy straight into LDS (issued into buf passed separately)
        } else {
            #pragma unroll
            for (int i = 0; i < 2; ++i) {
                int g = tid + i * 256, r = g >> 3, c4 = (g & 7) * 4;
                aRegF[i] = *(const v4f*)&Af[(size_t)(rowblk + r) * CD + k0 + c4];
            }
        }
    };
    auto loadB = [&](int k0) {
        #pragma unroll
        for (int i = 0; i < 8; ++i) {
            int p = tid + i * 256, cc = p & 127, kk = (p >> 7) * 2;
            bReg0[i] = W[(size_t)(k0 + kk) * CD + colblk + cc];
            bReg1[i] = W[(size_t)(k0 + kk + 1) * CD + colblk + cc];
        }
    };
    auto asyncA = [&](int buf, int k0) {
        unsigned la = lds_addr_of(&sA[buf][ar * 48 + ac8]);
        async_copy_b128(la, &Ah[(size_t)(rowblk + ar) * CD + k0 + ac8]);
    };
    auto storeTile = [&](int buf) {
        if constexpr (!AISHALF) {
            #pragma unroll
            for (int i = 0; i < 2; ++i) {
                int g = tid + i * 256, r = g >> 3, c4 = (g & 7) * 4;
                v4h hv;
                hv[0] = (_Float16)aRegF[i][0]; hv[1] = (_Float16)aRegF[i][1];
                hv[2] = (_Float16)aRegF[i][2]; hv[3] = (_Float16)aRegF[i][3];
                *(v4h*)&sA[buf][r * 48 + c4] = hv;
            }
        }
        #pragma unroll
        for (int i = 0; i < 8; ++i) {
            int p = tid + i * 256, cc = p & 127, kk = (p >> 7) * 2;
            v2h hv; hv[0] = (_Float16)bReg0[i]; hv[1] = (_Float16)bReg1[i];
            *(v2h*)&sB[buf][cc * 48 + kk] = hv;
        }
    };

    // ---- prologue: stage tile 0 into buffer 0 ----
    if constexpr (AISHALF) asyncA(0, 0);
    loadA(0);
    loadB(0);
    storeTile(0);

    const int NIT = CD / 32;   // 32
    int cur = 0;
    for (int kt = 0; kt < NIT; ++kt) {
        if constexpr (AISHALF) wait_asynccnt0();
        __syncthreads();

        const int k1 = (kt + 1) * 32;
        if (kt + 1 < NIT) {
            if constexpr (AISHALF) asyncA(cur ^ 1, k1);
            loadA(k1);
            loadB(k1);
        }

        // ---- compute from buffer `cur` ----
        // A fragment (16x32 f16): lanes 0-15 K 0-15, lanes 16-31 K 16-31 (contiguous)
        const _Float16* ap = &sA[cur][(rt * 16 + llo) * 48 + lhi * 16];
        v8h a0 = *(const v8h*)ap;
        v8h a1 = *(const v8h*)(ap + 8);
        v16h afrag = make_v16h(a0, a1);
        #pragma unroll
        for (int t = 0; t < 4; ++t) {
            // B fragment (32x16): lane col = llo, K = lhi*16 + j (contiguous)
            const _Float16* bp = &sB[cur][(cbase + t * 16 + llo) * 48 + lhi * 16];
            v16h bfrag = *(const v16h*)bp;
            acc[t] = wmma_f16(afrag, bfrag, acc[t]);
        }

        if (kt + 1 < NIT) storeTile(cur ^ 1);
        cur ^= 1;
    }

    // Epilogue: C layout -> row = i + 8*lhi, col = tile + llo
    #pragma unroll
    for (int t = 0; t < 4; ++t) {
        #pragma unroll
        for (int i = 0; i < 8; ++i) {
            int row = rowblk + rt * 16 + i + lhi * 8;
            int col = colblk + cbase + t * 16 + llo;
            float v = acc[t][i] + bias[col];
            int b = row >> 10, n = row & 1023;
            int h = col >> 6,  d = col & 63;
            if constexpr (MODE == 0) {
                int s = (col >> 3) & 7, p = s >> 1;
                float coef = COEF_BASE;
                if (s & 1) coef = -coef * __expf(lambda_log[h * 4 + p]);
                v *= coef;
                dstH[((size_t)(b * NH + h) * 1024 + n) * HDIM + d] = (_Float16)v;
            } else if constexpr (MODE == 1) {
                dstH[((size_t)(b * NH + h) * 1024 + n) * HDIM + d] = (_Float16)v;
            } else if constexpr (MODE == 2) {
                dstH[((size_t)(b * NH + h) * HDIM + d) * 1024 + n] = (_Float16)v;
            } else {
                dstF[(size_t)row * CD + col] = v;
            }
        }
    }
}

// ---------------------------------------------------------------------------
// Attention kernel: one workgroup per (b, h, 16-row block of N).
// Phase 1: logits strip 16x1024 (wave w owns cols [w*128, w*128+128)), WMMA K=64.
// Phase 2: cross-wave softmax (shfl width-16 + small LDS), probs f32 -> d_out,
//          f16 -> LDS.
// Phase 3: waves 0..3 compute attn_out 16x64 = probs @ V (K=1024, 32 WMMAs).
// ---------------------------------------------------------------------------
__global__ __launch_bounds__(256)
void idattn_attn_kernel(const _Float16* __restrict__ Qt,
                        const _Float16* __restrict__ Kt,
                        const _Float16* __restrict__ Vt,
                        float* __restrict__ probs,
                        _Float16* __restrict__ attn_out)
{
    __shared__ alignas(32) _Float16 sprob[16 * 1032];   // 16 rows x 1024 probs (f16)
    __shared__ float pmax[8][16];
    __shared__ float psum[8][16];

    const int tid  = threadIdx.x;
    const int wave = tid >> 5;
    const int lane = tid & 31;
    const int lhi  = lane >> 4;
    const int llo  = lane & 15;

    const int nb = blockIdx.x;          // 0..63
    const int h  = blockIdx.y;          // 0..15
    const int b  = blockIdx.z;          // 0..1
    const int bh = b * NH + h;
    const int n0 = nb * 16;

    const _Float16* qbase = Qt + ((size_t)bh * 1024 + n0) * HDIM;
    const _Float16* kbase = Kt + (size_t)bh * 1024 * HDIM;
    const _Float16* vbase = Vt + (size_t)bh * HDIM * 1024;

    const int colw = wave * 128;
    v8f acc[8] = {};

    // ---- Phase 1: logits = Q' K^T over d=64 (2 chunks of 32) ----
    #pragma unroll
    for (int kc = 0; kc < 2; ++kc) {
        const _Float16* ap = qbase + llo * HDIM + kc * 32 + lhi * 16;
        v8h a0 = *(const v8h*)ap;
        v8h a1 = *(const v8h*)(ap + 8);
        v16h af = make_v16h(a0, a1);
        #pragma unroll
        for (int t = 0; t < 8; ++t) {
            const _Float16* bp = kbase + (size_t)(colw + t * 16 + llo) * HDIM + kc * 32 + lhi * 16;
            v16h bf = *(const v16h*)bp;
            acc[t] = wmma_f16(af, bf, acc[t]);
        }
    }

    // ---- Phase 2: softmax over the 1024 columns ----
    float rmax[8], rsum[8];
    #pragma unroll
    for (int i = 0; i < 8; ++i) {
        float m = acc[0][i];
        #pragma unroll
        for (int t = 1; t < 8; ++t) m = fmaxf(m, acc[t][i]);
        #pragma unroll
        for (int off = 1; off < 16; off <<= 1) m = fmaxf(m, __shfl_xor(m, off, 16));
        rmax[i] = m;
    }
    if (llo == 0) {
        #pragma unroll
        for (int i = 0; i < 8; ++i) pmax[wave][i + lhi * 8] = rmax[i];
    }
    __syncthreads();
    #pragma unroll
    for (int i = 0; i < 8; ++i) {
        float m = pmax[0][i + lhi * 8];
        #pragma unroll
        for (int w = 1; w < 8; ++w) m = fmaxf(m, pmax[w][i + lhi * 8]);
        rmax[i] = m;
    }
    #pragma unroll
    for (int i = 0; i < 8; ++i) {
        float s = 0.f;
        #pragma unroll
        for (int t = 0; t < 8; ++t) {
            float e = __expf(acc[t][i] - rmax[i]);
            acc[t][i] = e;
            s += e;
        }
        #pragma unroll
        for (int off = 1; off < 16; off <<= 1) s += __shfl_xor(s, off, 16);
        rsum[i] = s;
    }
    if (llo == 0) {
        #pragma unroll
        for (int i = 0; i < 8; ++i) psum[wave][i + lhi * 8] = rsum[i];
    }
    __syncthreads();
    float* prow = probs + ((size_t)bh * 1024 + n0) * 1024;
    #pragma unroll
    for (int i = 0; i < 8; ++i) {
        float s = psum[0][i + lhi * 8];
        #pragma unroll
        for (int w = 1; w < 8; ++w) s += psum[w][i + lhi * 8];
        float inv = 1.0f / s;
        int r = i + lhi * 8;
        #pragma unroll
        for (int t = 0; t < 8; ++t) {
            float p = acc[t][i] * inv;
            int c = colw + t * 16 + llo;
            prow[(size_t)r * 1024 + c] = p;            // attn_probs output (f32)
            sprob[r * 1032 + c] = (_Float16)p;         // f16 copy for PV
        }
    }
    __syncthreads();

    // ---- Phase 3: attn_out = probs @ V  (waves 0..3, one 16x16 d-tile each) ----
    if (wave < 4) {
        const int dcol = wave * 16;
        v8f oacc = {};
        #pragma unroll 8
        for (int kc = 0; kc < 32; ++kc) {
            const _Float16* ap = &sprob[llo * 1032 + kc * 32 + lhi * 16];
            v8h a0 = *(const v8h*)ap;
            v8h a1 = *(const v8h*)(ap + 8);
            v16h af = make_v16h(a0, a1);
            const _Float16* bp = vbase + (size_t)(dcol + llo) * 1024 + kc * 32 + lhi * 16;
            v16h bf = *(const v16h*)bp;
            oacc = wmma_f16(af, bf, oacc);
        }
        #pragma unroll
        for (int i = 0; i < 8; ++i) {
            int n = n0 + i + lhi * 8;
            int c = h * HDIM + dcol + llo;
            attn_out[(size_t)(b * 1024 + n) * CD + c] = (_Float16)oacc[i];
        }
    }
}

// ---------------------------------------------------------------------------
extern "C" void kernel_launch(void* const* d_in, const int* in_sizes, int n_in,
                              void* d_out, int out_size, void* d_ws, size_t ws_size,
                              hipStream_t stream) {
    const float* query      = (const float*)d_in[0];
    const float* key        = (const float*)d_in[1];
    const float* value      = (const float*)d_in[2];
    const float* Wq         = (const float*)d_in[3];
    const float* bq         = (const float*)d_in[4];
    const float* Wk         = (const float*)d_in[5];
    const float* bk         = (const float*)d_in[6];
    const float* Wv         = (const float*)d_in[7];
    const float* bv         = (const float*)d_in[8];
    const float* Wo         = (const float*)d_in[9];
    const float* bo         = (const float*)d_in[10];
    const float* lambda_log = (const float*)d_in[11];

    // Workspace layout (f16): Qt | Kt | Vt | attn_out, each 2*16*1024*64 = 2M halfs
    const size_t SEG = (size_t)2 * NH * 1024 * HDIM;   // 2,097,152 halfs
    _Float16* Qt = (_Float16*)d_ws;
    _Float16* Kt = Qt + SEG;
    _Float16* Vt = Kt + SEG;
    _Float16* AO = Vt + SEG;

    float* out_proj = (float*)d_out;                       // [2,1024,1024]
    float* probs    = out_proj + (size_t)2 * 1024 * CD;    // [2,16,1024,1024]

    dim3 ggrid(CD / 128, ROWS_TOT / 64);   // (8, 32)
    dim3 gblock(256);

    // Q' = (query @ Wq + bq) * c[h,s]   -> head-major f16
    idattn_gemm_kernel<0, false><<<ggrid, gblock, 0, stream>>>(query, Wq, bq, lambda_log, Qt, nullptr);
    // K = key @ Wk + bk                 -> head-major f16
    idattn_gemm_kernel<1, false><<<ggrid, gblock, 0, stream>>>(key, Wk, bk, nullptr, Kt, nullptr);
    // V = value @ Wv + bv               -> transposed [bh][d][m] f16
    idattn_gemm_kernel<2, false><<<ggrid, gblock, 0, stream>>>(value, Wv, bv, nullptr, Vt, nullptr);

    // softmax(Q'K^T), probs -> d_out, attn_out = probs @ V -> ws
    idattn_attn_kernel<<<dim3(64, NH, 2), gblock, 0, stream>>>(Qt, Kt, Vt, probs, AO);

    // output = attn_out @ Wo + bo       -> d_out f32 (A staged via async-to-LDS)
    idattn_gemm_kernel<3, true><<<ggrid, gblock, 0, stream>>>(AO, Wo, bo, nullptr, nullptr, out_proj);
}
